// Network_81810537054229
// MI455X (gfx1250) — matrix operand order repaired and checked by reading, compiled, not run on hardware
//
#include <hip/hip_runtime.h>
#include <hip/hip_bf16.h>
#include <math.h>

// CDNA5 WMMA vector types (probe-confirmed signatures)
typedef __attribute__((ext_vector_type(16))) __bf16 v16bf;
typedef __attribute__((ext_vector_type(8)))  float  v8f;

#define B_  4096
#define F_  39
#define V_  100000
#define D_  10

// ---------------------------------------------------------------------------
// Convert W[j][h][f] (f32) into bf16 A-fragments pre-swizzled for
// v_wmma_f32_16x16x32_bf16. A layout used here (16x32, MxK, contiguous K):
//   lanes 0-15 : M = lane,     K = {0..15} (t = K)
//   lanes 16-31: M = lane-16,  K = {16..31} (t = K-16)
// Fragment = 512 bf16, stored lane-major: frag[lane*16 + t].
// ---------------------------------------------------------------------------
__global__ void prep_weights_kernel(const float* __restrict__ W,
                                    __bf16* __restrict__ dst,
                                    int J, int KTOT, int KTILES, int MTILES) {
  int tid = blockIdx.x * blockDim.x + threadIdx.x;
  int total = MTILES * KTILES * 512;
  if (tid >= total) return;
  int frag = tid >> 9;
  int lane = (tid >> 4) & 31;
  int t    = tid & 15;
  int mtile = frag / KTILES;
  int ktile = frag - mtile * KTILES;
  int m  = lane & 15;
  int hh = lane >> 4;
  int kin = t + 16 * hh;   // K index within the 32-chunk
  int j  = mtile * 16 + m;
  int kk = ktile * 32 + kin;
  float v = 0.f;
  if (j < J && kk < KTOT) v = W[(size_t)j * KTOT + kk];
  dst[tid] = (__bf16)v;
}

// ---------------------------------------------------------------------------
// x0t[d][f][b] = emb[f, xi[b,f], d] * xv[b,f]   (b contiguous -> WMMA columns)
// ---------------------------------------------------------------------------
__global__ void embed_kernel(const int* __restrict__ xi,
                             const float* __restrict__ xv,
                             const float* __restrict__ emb,
                             float* __restrict__ x0) {
  int b = blockIdx.x * blockDim.x + threadIdx.x;
  int f = blockIdx.y;
  int d = blockIdx.z;
  if (b >= B_) return;
  int idx = xi[b * F_ + f];
  float v = emb[((size_t)f * V_ + idx) * D_ + d] * xv[b * F_ + f];
  x0[((size_t)d * F_ + f) * B_ + b] = v;
}

// ---------------------------------------------------------------------------
// Fused CIN layer: hout[d][j][b] = bias[j] + sum_k W[j,k] * z[k,(d,b)]
// with k = h*F_+f, z = hin[d][h][b]*x0[d][f][b] built per 32x16 fragment in
// registers from LDS tiles (z never touches memory; layer-1 z would be 640MB).
//
// - Zero-padded LDS rows + OOB table entries -> branch-free inner loop.
// - Table entries are packed element offsets ((h_row<<16)|x_row), loaded as
//   uint4 (LDS broadcast: wave-uniform per lane-half).
// - Each wave owns NT=2 column tiles: every A fragment load feeds 2 WMMAs,
//   halving weight traffic from L2.
// ---------------------------------------------------------------------------
template <int HIN, int J, int MTILES>
__global__ __launch_bounds__(64) void cin_layer_kernel(
    const float* __restrict__ hin, const float* __restrict__ x0,
    const __bf16* __restrict__ wbf, const float* __restrict__ bias,
    float* __restrict__ hout) {
  constexpr int KTOT   = HIN * F_;
  constexpr int KTILES = (KTOT + 31) / 32;
  constexpr int WAVES  = 2;
  constexpr int NT     = 2;            // 16-column tiles per wave
  constexpr int NCOL   = 16 * NT;      // 32 columns per wave
  constexpr int ZROW_H = HIN;          // zeroed row terminating hin block
  constexpr int XBASE  = HIN + 1;      // first x0 row
  constexpr int ZROW_X = XBASE + F_;   // zeroed row terminating x0 block
  constexpr int ROWS   = HIN + F_ + 2;

  __shared__ unsigned s_tab[KTILES * 32];        // k -> packed row offsets
  __shared__ float    s_tile[WAVES][ROWS * NCOL];

  const int tid  = threadIdx.x;
  const int w    = tid >> 5;
  const int lane = tid & 31;

  const int nb   = B_ / (NCOL * WAVES);
  const int d    = blockIdx.x / nb;
  const int bblk = blockIdx.x - d * nb;
  const int b0   = (bblk * WAVES + w) * NCOL;

  // k -> (h,f) table as packed LDS element offsets; OOB k -> zero rows.
  for (int kk = tid; kk < KTILES * 32; kk += 64) {
    unsigned e;
    if (kk < KTOT) {
      int h = kk / F_;
      int f = kk - h * F_;
      e = ((unsigned)(h * NCOL) << 16) | (unsigned)((XBASE + f) * NCOL);
    } else {
      e = ((unsigned)(ZROW_H * NCOL) << 16) | (unsigned)(ZROW_X * NCOL);
    }
    s_tab[kk] = e;
  }
  // Stage [hin | 0 | x0 | 0] tiles, NCOL columns per wave.
  for (int i = lane; i < ROWS * NCOL; i += 32) {
    int row = i / NCOL, n = i % NCOL;
    float v = 0.f;
    if (row < HIN)
      v = hin[((size_t)d * HIN + row) * B_ + b0 + n];
    else if (row >= XBASE && row < ZROW_X)
      v = x0[((size_t)d * F_ + (row - XBASE)) * B_ + b0 + n];
    s_tile[w][i] = v;
  }
  __syncthreads();

  const float* base = &s_tile[w][0];

  v8f acc[NT][MTILES];
#pragma unroll
  for (int nt = 0; nt < NT; ++nt)
#pragma unroll
    for (int m = 0; m < MTILES; ++m)
#pragma unroll
      for (int r = 0; r < 8; ++r) acc[nt][m][r] = 0.f;

  const int n  = lane & 15;   // column within a 16-tile
  const int hh = lane >> 4;   // K half for B operand / M half for C/D

  for (int kt = 0; kt < KTILES; ++kt) {
    // 16 table entries for this lane-half as 4x uint4 (LDS broadcast loads)
    const uint4* tp =
        reinterpret_cast<const uint4*>(&s_tab[kt * 32 + hh * 16]);
    uint4 tq0 = tp[0], tq1 = tp[1], tq2 = tp[2], tq3 = tp[3];
    unsigned ent[16] = {tq0.x, tq0.y, tq0.z, tq0.w, tq1.x, tq1.y, tq1.z, tq1.w,
                        tq2.x, tq2.y, tq2.z, tq2.w, tq3.x, tq3.y, tq3.z, tq3.w};

    // Build B operands (32x16 z fragments): lane = column, lane-half = K half
    v16bf bf[NT];
#pragma unroll
    for (int nt = 0; nt < NT; ++nt) {
      const int nn = nt * 16 + n;
#pragma unroll
      for (int t = 0; t < 16; ++t) {
        unsigned e = ent[t];
        float z = base[(e >> 16) + nn] * base[(e & 0xffffu) + nn];
        bf[nt][t] = (__bf16)z;
      }
    }
    // One A fragment per M tile feeds NT WMMAs
#pragma unroll
    for (int m = 0; m < MTILES; ++m) {
      const v16bf a = *reinterpret_cast<const v16bf*>(
          wbf + ((size_t)(m * KTILES + kt) << 9) + lane * 16);
#pragma unroll
      for (int nt = 0; nt < NT; ++nt)
        acc[nt][m] = __builtin_amdgcn_wmma_f32_16x16x32_bf16(
            false, a, false, bf[nt], (short)0, acc[nt][m], false, false);
    }
  }

  // C/D layout: lanes 0-15 -> rows 0-7, lanes 16-31 -> rows 8-15
#pragma unroll
  for (int nt = 0; nt < NT; ++nt) {
#pragma unroll
    for (int m = 0; m < MTILES; ++m) {
#pragma unroll
      for (int r = 0; r < 8; ++r) {
        int j = m * 16 + r + 8 * hh;
        if (j < J)
          hout[((size_t)d * J + j) * B_ + b0 + nt * 16 + n] =
              acc[nt][m][r] + bias[j];
      }
    }
  }
}

// ---------------------------------------------------------------------------
// Sum-pool over D, concat [100|100|50], dot with out_w, sigmoid.
// ---------------------------------------------------------------------------
__global__ void head_kernel(const float* __restrict__ h1,
                            const float* __restrict__ h2,
                            const float* __restrict__ h3,
                            const float* __restrict__ out_w,
                            const float* __restrict__ out_b,
                            float* __restrict__ out) {
  int b = blockIdx.x * blockDim.x + threadIdx.x;
  if (b >= B_) return;
  float acc = out_b[0];
  for (int j = 0; j < 100; ++j) {
    float s = 0.f;
    for (int d = 0; d < D_; ++d) s += h1[((size_t)d * 100 + j) * B_ + b];
    acc += s * out_w[j];
  }
  for (int j = 0; j < 100; ++j) {
    float s = 0.f;
    for (int d = 0; d < D_; ++d) s += h2[((size_t)d * 100 + j) * B_ + b];
    acc += s * out_w[100 + j];
  }
  for (int j = 0; j < 50; ++j) {
    float s = 0.f;
    for (int d = 0; d < D_; ++d) s += h3[((size_t)d * 50 + j) * B_ + b];
    acc += s * out_w[200 + j];
  }
  out[b] = 1.f / (1.f + expf(-acc));
}

// ---------------------------------------------------------------------------
extern "C" void kernel_launch(void* const* d_in, const int* in_sizes, int n_in,
                              void* d_out, int out_size, void* d_ws,
                              size_t ws_size, hipStream_t stream) {
  const int*   xi   = (const int*)  d_in[0];
  const float* xv   = (const float*)d_in[1];
  const float* emb  = (const float*)d_in[2];
  const float* W0   = (const float*)d_in[3];
  const float* b0   = (const float*)d_in[4];
  const float* W1   = (const float*)d_in[5];
  const float* b1   = (const float*)d_in[6];
  const float* W2   = (const float*)d_in[7];
  const float* b2   = (const float*)d_in[8];
  const float* outw = (const float*)d_in[9];
  const float* outb = (const float*)d_in[10];
  float* out = (float*)d_out;
  (void)in_sizes; (void)n_in; (void)out_size; (void)ws_size;

  // Workspace carve-out (~49 MB): all regions fully rewritten every call.
  char* ws = (char*)d_ws;
  size_t off = 0;
  auto carve = [&](size_t bytes) -> void* {
    void* p = ws + off;
    off = (off + bytes + 255) & ~(size_t)255;
    return p;
  };
  float*  x0  = (float*) carve((size_t)D_ * F_  * B_ * 4);
  float*  h1  = (float*) carve((size_t)D_ * 100 * B_ * 4);
  float*  h2  = (float*) carve((size_t)D_ * 100 * B_ * 4);
  float*  h3  = (float*) carve((size_t)D_ * 50  * B_ * 4);
  __bf16* w0b = (__bf16*)carve((size_t)7 * 48  * 512 * 2);
  __bf16* w1b = (__bf16*)carve((size_t)7 * 122 * 512 * 2);
  __bf16* w2b = (__bf16*)carve((size_t)4 * 122 * 512 * 2);

  // 1) bf16 weight fragments (tiny; same stream, runs before embedding gather)
  int t0 = 7 * 48  * 512;
  int t1 = 7 * 122 * 512;
  int t2 = 4 * 122 * 512;
  prep_weights_kernel<<<(t0 + 255) / 256, 256, 0, stream>>>(W0, w0b, 100, 39 * 39,  48, 7);
  prep_weights_kernel<<<(t1 + 255) / 256, 256, 0, stream>>>(W1, w1b, 100, 100 * 39, 122, 7);
  prep_weights_kernel<<<(t2 + 255) / 256, 256, 0, stream>>>(W2, w2b, 50,  100 * 39, 122, 4);

  // 2) embedding gather -> x0t[d][f][b]
  embed_kernel<<<dim3(B_ / 256, F_, D_), 256, 0, stream>>>(xi, xv, emb, x0);

  // 3) three fused CIN layers (WMMA); 2 waves/block, 32 columns/wave
  int grid = D_ * (B_ / 64);
  cin_layer_kernel<39, 100, 7><<<grid, 64, 0, stream>>>(x0, x0, w0b, b0, h1);
  cin_layer_kernel<100, 100, 7><<<grid, 64, 0, stream>>>(h1, x0, w1b, b1, h2);
  cin_layer_kernel<100, 50, 4><<<grid, 64, 0, stream>>>(h2, x0, w2b, b2, h3);

  // 4) pool + linear + sigmoid
  head_kernel<<<(B_ + 255) / 256, 256, 0, stream>>>(h1, h2, h3, outw, outb, out);
}
